// FastAttention_2362232013124
// MI455X (gfx1250) — compile-verified, matrix operand order checked by reading, not run on hardware
//
#include <hip/hip_runtime.h>
#include <math.h>

typedef __attribute__((ext_vector_type(16))) _Float16 v16h;
typedef __attribute__((ext_vector_type(8)))  float    v8f;

#define B_   4
#define H_   16
#define NN   4096
#define DD   64
#define MM   266      // valid features
#define MP   288      // padded to multiple of 32
#define TN   128      // rows per block
#define BH   (B_*H_)

__constant__ const float kNORM  = 0.35355339059327373f;  // 64^-0.25
__constant__ const float kRATIO = 0.06131393394849658f;  // 266^-0.5
__constant__ const float kEPS   = 1e-4f;

// K index base for 16-bit A/B fragments (ISA 7.12.2):
// VGPR v (of 8), lane-half h: K = (v<4?0:16) + 8*h + 2*(v&3)
__device__ __forceinline__ int kidx(int v, int hlf) {
    return ((v >> 2) << 4) | (hlf << 3) | ((v & 3) << 1);
}

// A fragment: 16 rows (lane&15) x 32 K, row-major f32 source (global)
__device__ __forceinline__ v16h load_a_row_f32(const float* __restrict__ row,
                                               int kofs, int hlf) {
    v16h a;
#pragma unroll
    for (int v = 0; v < 8; ++v) {
        int k = kofs + kidx(v, hlf);
        a[2 * v]     = (_Float16)row[k];
        a[2 * v + 1] = (_Float16)row[k + 1];
    }
    return a;
}

// B fragment from LDS proj [MP][DD] (f16): B[K=d][n=m], col = mt*16 + lane&15
__device__ __forceinline__ v16h load_b_proj(const _Float16* __restrict__ projs,
                                            int mt, int kofs, int lane) {
    int m = mt * 16 + (lane & 15);
    int hlf = lane >> 4;
    const _Float16* p = projs + m * DD;
    v16h b;
#pragma unroll
    for (int v = 0; v < 8; ++v) {
        int k = kofs + kidx(v, hlf);
        b[2 * v]     = p[k];
        b[2 * v + 1] = p[k + 1];
    }
    return b;
}

// A fragment = kp^T: A[row=m][K=n] from LDS kp [TN][MP]
__device__ __forceinline__ v16h load_a_kpT(const _Float16* __restrict__ kp,
                                           int mt, int nofs, int lane) {
    int m = mt * 16 + (lane & 15);
    int hlf = lane >> 4;
    v16h a;
#pragma unroll
    for (int v = 0; v < 8; ++v) {
        int n = nofs + kidx(v, hlf);
        a[2 * v]     = kp[n * MP + m];
        a[2 * v + 1] = kp[(n + 1) * MP + m];
    }
    return a;
}

// B fragment from LDS v [TN][DD]: B[K=n][col=e]
__device__ __forceinline__ v16h load_b_vl(const _Float16* __restrict__ vl,
                                          int et, int nofs, int lane) {
    int e = et * 16 + (lane & 15);
    int hlf = lane >> 4;
    v16h b;
#pragma unroll
    for (int v = 0; v < 8; ++v) {
        int n = nofs + kidx(v, hlf);
        b[2 * v]     = vl[n * DD + e];
        b[2 * v + 1] = vl[(n + 1) * DD + e];
    }
    return b;
}

// A fragment from LDS qp [TN][MP]: A[row=n][K=m]
__device__ __forceinline__ v16h load_a_qp(const _Float16* __restrict__ qp,
                                          int nrow, int mofs, int hlf) {
    const _Float16* p = qp + nrow * MP;
    v16h a;
#pragma unroll
    for (int v = 0; v < 8; ++v) {
        int k = mofs + kidx(v, hlf);
        a[2 * v]     = p[k];
        a[2 * v + 1] = p[k + 1];
    }
    return a;
}

// B fragment from LDS ctx [MP][DD]: B[K=m][col=e]
__device__ __forceinline__ v16h load_b_ctx(const _Float16* __restrict__ ctxl,
                                           int et, int mofs, int lane) {
    int e = et * 16 + (lane & 15);
    int hlf = lane >> 4;
    v16h b;
#pragma unroll
    for (int v = 0; v < 8; ++v) {
        int m = mofs + kidx(v, hlf);
        b[2 * v]     = ctxl[m * DD + e];
        b[2 * v + 1] = ctxl[(m + 1) * DD + e];
    }
    return b;
}

__device__ __forceinline__ v8f wmma_f16(v16h a, v16h b, v8f c) {
    return __builtin_amdgcn_wmma_f32_16x16x32_f16(false, a, false, b,
                                                  (short)0, c, false, false);
}

__device__ __forceinline__ void atomicMaxF(float* addr, float val) {
    if (val >= 0.f) atomicMax((int*)addr, __float_as_int(val));
    else            atomicMin((unsigned int*)addr, __float_as_uint(val));
}

// ---------------- kernel 0: init workspace ----------------
__global__ void __launch_bounds__(256) perf_init(float* __restrict__ ws, int total) {
    for (int i = blockIdx.x * 256 + threadIdx.x; i < total; i += gridDim.x * 256)
        ws[i] = (i < BH) ? -1e30f : 0.f;
}

// ---------------- kernel 1: global max of k.proj^T per (b,h) ----------------
__global__ void __launch_bounds__(256) perf_kmax(const float* __restrict__ kin,
                                                 const float* __restrict__ proj,
                                                 float* __restrict__ mxk) {
    __shared__ _Float16 projs[MP * DD];
    int bh = blockIdx.y, chunk = blockIdx.x, tid = threadIdx.x;
    for (int i = tid; i < MP * DD; i += 256) {
        int m = i / DD, d = i % DD;
        projs[i] = (m < MM) ? (_Float16)(kNORM * proj[m * DD + d]) : (_Float16)0.f;
    }
    __syncthreads();

    int w = tid >> 5, lane = tid & 31, hlf = lane >> 4, ln = lane & 15;
    const float* krow = kin + (((size_t)bh * NN) + chunk * TN + w * 16 + ln) * DD;
    v16h a0 = load_a_row_f32(krow, 0, hlf);
    v16h a1 = load_a_row_f32(krow, 32, hlf);

    float mx = -1e30f;
    for (int mt = 0; mt < 17; ++mt) {   // mt=17 is fully padded: skip
        v8f c = {};
        c = wmma_f16(a0, load_b_proj(projs, mt, 0, lane), c);
        c = wmma_f16(a1, load_b_proj(projs, mt, 32, lane), c);
        if (mt * 16 + ln < MM) {
#pragma unroll
            for (int r = 0; r < 8; ++r) mx = fmaxf(mx, c[r]);
        }
    }
    for (int off = 16; off; off >>= 1) mx = fmaxf(mx, __shfl_xor(mx, off, 32));
    if (lane == 0) atomicMaxF(&mxk[bh], mx);
}

// ---------------- kernel 2: kp, context = kp^T @ v, k_cumsum ----------------
__global__ void __launch_bounds__(256) perf_ctx(const float* __restrict__ kin,
                                                const float* __restrict__ vin,
                                                const float* __restrict__ proj,
                                                const float* __restrict__ mxk,
                                                float* __restrict__ ksum,
                                                float* __restrict__ ctx) {
    __shared__ _Float16 projs[MP * DD];
    __shared__ _Float16 kp[TN * MP];
    __shared__ _Float16 vl[TN * DD];
    __shared__ float    diag[TN];
    int bh = blockIdx.y, chunk = blockIdx.x, tid = threadIdx.x;
    const float* kslab = kin + (((size_t)bh * NN) + chunk * TN) * DD;
    const float* vslab = vin + (((size_t)bh * NN) + chunk * TN) * DD;

    for (int i = tid; i < MP * DD; i += 256) {
        int m = i / DD, d = i % DD;
        projs[i] = (m < MM) ? (_Float16)(kNORM * proj[m * DD + d]) : (_Float16)0.f;
    }
    for (int i = tid; i < TN * DD; i += 256) vl[i] = (_Float16)vslab[i];
    if (tid < TN) {
        const float* r = kslab + tid * DD;
        float s = 0.f;
#pragma unroll 8
        for (int d = 0; d < DD; ++d) s += r[d] * r[d];
        diag[tid] = 0.0625f * s;   // 0.5 * d^-0.5 * ||x||^2
    }
    __syncthreads();

    float mx = mxk[bh];
    int w = tid >> 5, lane = tid & 31, hlf = lane >> 4, ln = lane & 15;
    const float* krow = kslab + (w * 16 + ln) * DD;
    v16h a0 = load_a_row_f32(krow, 0, hlf);
    v16h a1 = load_a_row_f32(krow, 32, hlf);

    for (int mt = 0; mt < 18; ++mt) {
        int m = mt * 16 + ln;
        if (mt < 17) {
            v8f c = {};
            c = wmma_f16(a0, load_b_proj(projs, mt, 0, lane), c);
            c = wmma_f16(a1, load_b_proj(projs, mt, 32, lane), c);
#pragma unroll
            for (int r = 0; r < 8; ++r) {
                int row = w * 16 + r + 8 * hlf;
                float kv = (m < MM)
                    ? kRATIO * (__expf(c[r] - diag[row] - mx) + kEPS) : 0.f;
                kp[row * MP + m] = (_Float16)kv;
            }
        } else {
#pragma unroll
            for (int r = 0; r < 8; ++r)
                kp[(w * 16 + r + 8 * hlf) * MP + m] = (_Float16)0.f;
        }
    }
    __syncthreads();

    // context tiles: 18 m-tiles x 4 e-tiles, round-robin over 8 waves
    for (int t = w; t < 18 * 4; t += 8) {
        int mt = t >> 2, et = t & 3;
        v8f c = {};
#pragma unroll
        for (int ks = 0; ks < 4; ++ks)
            c = wmma_f16(load_a_kpT(kp, mt, ks * 32, lane),
                         load_b_vl(vl, et, ks * 32, lane), c);
        int e = et * 16 + ln;
#pragma unroll
        for (int r = 0; r < 8; ++r) {
            int m = mt * 16 + r + 8 * hlf;
            atomicAdd(&ctx[(((size_t)bh * MP) + m) * DD + e], c[r]);
        }
    }
    // k_cumsum
    for (int m = tid; m < MP; m += 256) {
        float s = 0.f;
        for (int n = 0; n < TN; ++n) s += (float)kp[n * MP + m];
        atomicAdd(&ksum[(size_t)bh * MP + m], s);
    }
}

// ---------------- kernel 3: qp, D^-1, out = (qp @ ctx) * D^-1 ----------------
__global__ void __launch_bounds__(256) perf_out(const float* __restrict__ qin,
                                                const float* __restrict__ proj,
                                                const float* __restrict__ ksum,
                                                const float* __restrict__ ctx,
                                                float* __restrict__ out) {
    __shared__ _Float16 projs[MP * DD];
    __shared__ _Float16 qp[TN * MP];
    __shared__ _Float16 ctxl[MP * DD];
    __shared__ float    ksuml[MP];
    __shared__ float    diag[TN];
    __shared__ float    dinv[TN];
    int bh = blockIdx.y, chunk = blockIdx.x, tid = threadIdx.x;
    const float* qslab = qin + (((size_t)bh * NN) + chunk * TN) * DD;

    for (int i = tid; i < MP * DD; i += 256) {
        int m = i / DD, d = i % DD;
        projs[i] = (m < MM) ? (_Float16)(kNORM * proj[m * DD + d]) : (_Float16)0.f;
        ctxl[i]  = (_Float16)ctx[((size_t)bh * MP) * DD + i];
    }
    for (int i = tid; i < MP; i += 256) ksuml[i] = ksum[(size_t)bh * MP + i];
    if (tid < TN) {
        const float* r = qslab + tid * DD;
        float s = 0.f;
#pragma unroll 8
        for (int d = 0; d < DD; ++d) s += r[d] * r[d];
        diag[tid] = 0.0625f * s;
    }
    __syncthreads();

    int w = tid >> 5, lane = tid & 31, hlf = lane >> 4, ln = lane & 15;
    const float* qrow = qslab + (w * 16 + ln) * DD;
    v16h a0 = load_a_row_f32(qrow, 0, hlf);
    v16h a1 = load_a_row_f32(qrow, 32, hlf);

    // pass 1: per-row max over valid m
    float rmax[8];
#pragma unroll
    for (int r = 0; r < 8; ++r) rmax[r] = -1e30f;
    for (int mt = 0; mt < 17; ++mt) {
        v8f c = {};
        c = wmma_f16(a0, load_b_proj(projs, mt, 0, lane), c);
        c = wmma_f16(a1, load_b_proj(projs, mt, 32, lane), c);
        int m = mt * 16 + ln;
#pragma unroll
        for (int r = 0; r < 8; ++r) {
            float v = (m < MM) ? c[r] : -1e30f;
            // reduce across the 16 lanes of this half (same row set)
            v = fmaxf(v, __shfl_xor(v, 1, 32));
            v = fmaxf(v, __shfl_xor(v, 2, 32));
            v = fmaxf(v, __shfl_xor(v, 4, 32));
            v = fmaxf(v, __shfl_xor(v, 8, 32));
            rmax[r] = fmaxf(rmax[r], v);
        }
    }
    // pass 2: qp into LDS
    for (int mt = 0; mt < 18; ++mt) {
        int m = mt * 16 + ln;
        if (mt < 17) {
            v8f c = {};
            c = wmma_f16(a0, load_b_proj(projs, mt, 0, lane), c);
            c = wmma_f16(a1, load_b_proj(projs, mt, 32, lane), c);
#pragma unroll
            for (int r = 0; r < 8; ++r) {
                int row = w * 16 + r + 8 * hlf;
                float qv = (m < MM)
                    ? kRATIO * (__expf(c[r] - diag[row] - rmax[r]) + kEPS) : 0.f;
                qp[row * MP + m] = (_Float16)qv;
            }
        } else {
#pragma unroll
            for (int r = 0; r < 8; ++r)
                qp[(w * 16 + r + 8 * hlf) * MP + m] = (_Float16)0.f;
        }
    }
    __syncthreads();

    if (tid < TN) {
        float d = 0.f;
        for (int m = 0; m < MP; ++m) d += (float)qp[tid * MP + m] * ksuml[m];
        dinv[tid] = 1.f / ((d == 0.f) ? 1e-5f : d);
    }
    __syncthreads();

    // out tile: each wave owns its 16-row group x 4 e-tiles
    for (int et = 0; et < 4; ++et) {
        v8f c = {};
#pragma unroll
        for (int ms = 0; ms < MP / 32; ++ms)
            c = wmma_f16(load_a_qp(qp, w * 16 + ln, ms * 32, hlf),
                         load_b_ctx(ctxl, et, ms * 32, lane), c);
        int e = et * 16 + ln;
#pragma unroll
        for (int r = 0; r < 8; ++r) {
            int row = w * 16 + r + 8 * hlf;
            int ng = chunk * TN + row;
            out[(((size_t)bh * NN) + ng) * DD + e] = c[r] * dinv[row];
        }
    }
}

extern "C" void kernel_launch(void* const* d_in, const int* in_sizes, int n_in,
                              void* d_out, int out_size, void* d_ws, size_t ws_size,
                              hipStream_t stream) {
    const float* q    = (const float*)d_in[0];
    const float* k    = (const float*)d_in[1];
    const float* v    = (const float*)d_in[2];
    const float* proj = (const float*)d_in[3];
    float* out = (float*)d_out;

    float* ws   = (float*)d_ws;
    float* mxk  = ws;                    // [BH]
    float* ksum = ws + BH;               // [BH][MP]
    float* ctx  = ws + BH + BH * MP;     // [BH][MP][DD]
    int total = BH + BH * MP + BH * MP * DD;

    perf_init<<<(total + 255) / 256, 256, 0, stream>>>(ws, total);
    dim3 grid(NN / TN, BH);
    perf_kmax<<<grid, 256, 0, stream>>>(k, proj, mxk);
    perf_ctx<<<grid, 256, 0, stream>>>(k, v, proj, mxk, ksum, ctx);
    perf_out<<<grid, 256, 0, stream>>>(q, proj, ksum, ctx, out);
}
